// AtomEdgeInteraction_38027640438917
// MI455X (gfx1250) — matrix operand order repaired and checked by reading, not compiled, not run on hardware
//
#include <hip/hip_runtime.h>

// ---------------------------------------------------------------------------
// AtomEdgeInteraction fused pipeline for gfx1250 (MI455X), wave32 + WMMA bf16.
//
//   pass0: zero d_out (node sums) + workspace (counts, BN accumulators)
//   pass1: per-edge  u = relu(concat(x[row], edge_attr) @ W^T + b)
//          - GEMM via v_wmma_f32_16x16x32_bf16 (K=80 padded to 96, 3 steps)
//          - accumulate per-feature sum/sumsq for BatchNorm stats
//          - scatter-add u into d_out[col] (L2-resident atomics) + counts
//   pass2: s[f] = gamma*rsqrt(var+eps), t[f] = beta - mean*s   (64 threads)
//   pass3: out[n,f] = (s[f]*sum[n,f] + t[f]*cnt[n]) / max(cnt,1) + x[n,f]
// ---------------------------------------------------------------------------

typedef __attribute__((ext_vector_type(16))) __bf16        v16bf;
typedef __attribute__((ext_vector_type(8)))  float         v8f;
typedef __attribute__((ext_vector_type(4)))  unsigned int  u32x4;
typedef __attribute__((ext_vector_type(2)))  unsigned int  u32x2;

#define K_TOT      80   // IN_F + EDGE_F
#define LDS_STRIDE 104  // 96 (padded K) + 8 pad: 52-dword row stride, conflict-light
#define EPB        128  // edges per block (8 waves x 16 edges)

__device__ __forceinline__ unsigned short f2bf(float f) {
  unsigned int u = __float_as_uint(f);
  u += 0x7FFFu + ((u >> 16) & 1u);          // round-to-nearest-even
  return (unsigned short)(u >> 16);
}
__device__ __forceinline__ unsigned int pack2(float a, float b) {
  return (unsigned int)f2bf(a) | ((unsigned int)f2bf(b) << 16);
}

__global__ void zero_pass(float* __restrict__ out, int nOut,
                          float* __restrict__ ws,  int nWs) {
  int i = blockIdx.x * blockDim.x + threadIdx.x;
  if (i < nOut) out[i] = 0.0f;
  if (i < nWs)  ws[i]  = 0.0f;
}

__global__ __launch_bounds__(256) void edge_pass(
    const float* __restrict__ x,        // [N,64]
    const int*   __restrict__ ei,       // [2,E] (row, col)
    const float* __restrict__ ea,       // [E,16]
    const float* __restrict__ W,        // [64,80] row-major (out, in)
    const float* __restrict__ bias,     // [64]
    float* __restrict__ outSums,        // [N,64]  scatter-sum of relu(u)
    float* __restrict__ cnt,            // [N]
    float* __restrict__ bnsum,          // [64]
    float* __restrict__ bnsq,           // [64]
    int E)
{
  __shared__ __align__(16) unsigned short wl[64][LDS_STRIDE];      // B = W^T (bf16)
  __shared__ __align__(16) unsigned short comb[8][16][LDS_STRIDE]; // per-wave A tiles
  __shared__ int   cols_s[8][16];
  __shared__ float lbn[64], lbq[64];

  const int tid  = threadIdx.x;
  const int w    = tid >> 5;
  const int lane = tid & 31;
  const int egBase = blockIdx.x * EPB + w * 16;

  // ---- phase 0: zero LDS (W zero-pad region K=80..95 must be 0) ----
  for (int i = tid; i < 64 * LDS_STRIDE; i += 256)
    (&wl[0][0])[i] = 0;
  if (tid < 64) { lbn[tid] = 0.0f; lbq[tid] = 0.0f; }
  __syncthreads();

  // ---- phase 1a: stage W as bf16 into LDS ----
  for (int i = tid; i < 64 * K_TOT; i += 256)
    wl[i / K_TOT][i % K_TOT] = f2bf(W[i]);

  // ---- phase 1b: stage this wave's 16 edges: concat(x[row], edge_attr) ----
  const int  el    = lane >> 1;           // 2 lanes per edge
  const int  half  = lane & 1;
  const int  e     = egBase + el;
  const bool valid = (e < E);
  const int  eC    = valid ? e : (E - 1);
  const int  row   = ei[eC];

  if (half == 0) {
    int col = valid ? ei[E + eC] : 0;
    cols_s[w][el] = col;
    if (valid) atomicAdd(&cnt[col], 1.0f);
  } else {
    u32x4 z = {0u, 0u, 0u, 0u};           // zero-pad K=80..95
    *reinterpret_cast<u32x4*>(&comb[w][el][80]) = z;
    *reinterpret_cast<u32x4*>(&comb[w][el][88]) = z;
  }

  const float4* x4  = reinterpret_cast<const float4*>(x);
  const float4* ea4 = reinterpret_cast<const float4*>(ea);
#pragma unroll
  for (int cc = 0; cc < 10; ++cc) {       // 10 float4 chunks per half-edge
    int c = half * 10 + cc;               // chunk index in [0,20): 0..15 = x, 16..19 = edge_attr
    float4 v;
    if (!valid)      v = make_float4(0.f, 0.f, 0.f, 0.f);
    else if (c < 16) v = x4[(size_t)row * 16 + c];
    else             v = ea4[(size_t)e * 4 + (c - 16)];
    u32x2 p;
    p.x = pack2(v.x, v.y);
    p.y = pack2(v.z, v.w);
    *reinterpret_cast<u32x2*>(&comb[w][el][c * 4]) = p;
  }
  __syncthreads();

  // ---- phase 2: GEMM, 3 K-steps x 4 N-tiles of v_wmma_f32_16x16x32_bf16 ----
  v8f acc[4] = {};
  const int mrow = lane & 15;
  const int aoff = (lane >> 4) * 8;       // A: lanes 0-15 hold K {0..7,16..23}, hi lanes +8
  const int boff = (lane >> 4) * 8;       // B: lanes 0-15 hold K {0..7,16..23}, hi lanes +8
#pragma unroll
  for (int ks = 0; ks < 3; ++ks) {
    const int kb = ks * 32;
    union { u32x4 q[2]; v16bf v; } af;
    af.q[0] = *reinterpret_cast<const u32x4*>(&comb[w][mrow][kb + aoff]);
    af.q[1] = *reinterpret_cast<const u32x4*>(&comb[w][mrow][kb + aoff + 16]);
#pragma unroll
    for (int j = 0; j < 4; ++j) {
      union { u32x4 q[2]; v16bf v; } bf;
      const unsigned short* bp = &wl[j * 16 + mrow][kb + boff];
      bf.q[0] = *reinterpret_cast<const u32x4*>(bp);
      bf.q[1] = *reinterpret_cast<const u32x4*>(bp + 16);
      acc[j] = __builtin_amdgcn_wmma_f32_16x16x32_bf16(
          false, af.v, false, bf.v, (short)0, acc[j], false, false);
    }
  }

  // ---- phase 3: bias + relu, BN partial sums, scatter-add ----
  const bool hi = (lane >= 16);
#pragma unroll
  for (int j = 0; j < 4; ++j) {
    const int   nj = j * 16 + mrow;       // feature index this lane owns in tile j
    const float bv = bias[nj];
    float bs = 0.0f, bq = 0.0f;
#pragma unroll
    for (int i = 0; i < 8; ++i) {
      const int m  = hi ? (i + 8) : i;    // local edge row per C-matrix layout
      const int ee = egBase + m;
      float v = acc[j][i] + bv;
      v = fmaxf(v, 0.0f);
      if (ee < E) {
        bs += v;
        bq += v * v;
        atomicAdd(&outSums[(size_t)cols_s[w][m] * 64 + nj], v);
      }
    }
    // lanes L and L^16 own the same feature -> pair-reduce, then LDS atomic
    bs += __shfl_xor(bs, 16, 32);
    bq += __shfl_xor(bq, 16, 32);
    if (lane < 16) {
      atomicAdd(&lbn[nj], bs);
      atomicAdd(&lbq[nj], bq);
    }
  }
  __syncthreads();

  // ---- phase 4: one global atomic per feature per block for BN stats ----
  if (tid < 64) {
    atomicAdd(&bnsum[tid], lbn[tid]);
    atomicAdd(&bnsq[tid],  lbq[tid]);
  }
}

__global__ void bn_params(const float* __restrict__ bnsum,
                          const float* __restrict__ bnsq,
                          const float* __restrict__ gamma,
                          const float* __restrict__ beta,
                          float* __restrict__ sArr,
                          float* __restrict__ tArr,
                          float invE) {
  int f = threadIdx.x;
  float mean = bnsum[f] * invE;
  float var  = fmaxf(bnsq[f] * invE - mean * mean, 0.0f);
  float s    = gamma[f] * rsqrtf(var + 1e-5f);
  sArr[f] = s;
  tArr[f] = beta[f] - mean * s;
}

__global__ void finalize_pass(float* __restrict__ out,
                              const float* __restrict__ x,
                              const float* __restrict__ cnt,
                              const float* __restrict__ sArr,
                              const float* __restrict__ tArr,
                              int total) {
  int i = blockIdx.x * blockDim.x + threadIdx.x;
  if (i >= total) return;
  int f = i & 63;
  int n = i >> 6;
  float c     = cnt[n];
  float denom = fmaxf(c, 1.0f);
  out[i] = (sArr[f] * out[i] + tArr[f] * c) / denom + x[i];
}

extern "C" void kernel_launch(void* const* d_in, const int* in_sizes, int n_in,
                              void* d_out, int out_size, void* d_ws, size_t ws_size,
                              hipStream_t stream) {
  const float* x     = (const float*)d_in[0];
  const int*   ei    = (const int*)  d_in[1];   // edge_index [2,E]
  const float* ea    = (const float*)d_in[2];
  const float* W     = (const float*)d_in[3];
  const float* b     = (const float*)d_in[4];
  const float* gamma = (const float*)d_in[5];
  const float* beta  = (const float*)d_in[6];
  float* out = (float*)d_out;
  float* ws  = (float*)d_ws;

  const int E = in_sizes[2] / 16;   // edge_attr [E,16]
  const int N = out_size / 64;

  // workspace layout: bnsum[64] | bnsq[64] | s[64] | t[64] | cnt[N]
  float* bnsum = ws;
  float* bnsq  = ws + 64;
  float* sArr  = ws + 128;
  float* tArr  = ws + 192;
  float* cntp  = ws + 256;
  const int nWs = 256 + N;

  const int zmax = (out_size > nWs) ? out_size : nWs;
  zero_pass<<<(zmax + 255) / 256, 256, 0, stream>>>(out, out_size, ws, nWs);

  edge_pass<<<(E + EPB - 1) / EPB, 256, 0, stream>>>(
      x, ei, ea, W, b, out, cntp, bnsum, bnsq, E);

  bn_params<<<1, 64, 0, stream>>>(bnsum, bnsq, gamma, beta, sArr, tArr,
                                  1.0f / (float)E);

  finalize_pass<<<(out_size + 255) / 256, 256, 0, stream>>>(
      out, x, cntp, sArr, tArr, out_size);
}
